// vit_80444737454956
// MI455X (gfx1250) — compile-verified, matrix operand order checked by reading, not run on hardware
//
#include <hip/hip_runtime.h>
#include <hip/hip_bf16.h>

// ---------------------------------------------------------------------------
// Types for CDNA5 WMMA (wave32): 16x16x32 bf16 -> f32
// ---------------------------------------------------------------------------
typedef __bf16 bf16x16 __attribute__((ext_vector_type(16)));
typedef float  f32x8   __attribute__((ext_vector_type(8)));

union ABFrag { bf16x16 v; unsigned int u32[8]; };

__device__ __forceinline__ unsigned short f2bf(float f) {
    union { float f; unsigned u; } c; c.f = f;
    unsigned r = c.u + 0x7FFFu + ((c.u >> 16) & 1u);   // round-to-nearest-even
    return (unsigned short)(r >> 16);
}
__device__ __forceinline__ unsigned int pk2(float lo, float hi) {
    return (unsigned)f2bf(lo) | ((unsigned)f2bf(hi) << 16);
}

enum { FL_TB = 1, FL_BIASN = 2, FL_BIASM = 4, FL_RELU = 8, FL_GELU = 16, FL_RES = 32 };

// ---------------------------------------------------------------------------
// Generic batched WMMA GEMM:  D = epi( A(MxK) * B(KxN) )
//  - block tile 128x64, BK=32, 8 waves (wave32); each wave computes a 32x32
//    sub-tile = four v_wmma_f32_16x16x32_bf16 per K-step.
//  - fp32 inputs converted to bf16 and packed 2-per-u32 in LDS, in the
//    fragment-native layout (row = M or N, col = K-pair) so every fragment is
//    8 aligned ds_load_b32.
//  - uniform full-tile fast path (no divergent guards) when the tile is
//    fully in-bounds; guarded path only on edge blocks.
//  - two-level batch (outer*inner) strides; epilogue: +bias(n|m), relu/gelu,
//    *alpha(device scalar), +Cres.
// ---------------------------------------------------------------------------
__global__ __launch_bounds__(256) void k_gemm(
    const float* __restrict__ A, const float* __restrict__ B,
    const float* __restrict__ bias, const float* __restrict__ Cres,
    float* __restrict__ D, const float* __restrict__ alpha_ptr,
    int M, int N, int K,
    long long lda, long long ldb, long long ldc, long long ldd,
    long long sAo, long long sAi, long long sBo, long long sBi,
    long long sCo, long long sCi, long long sDo, long long sDi,
    int innerN, int flags)
{
    __shared__ unsigned int Au[128][17];   // 128 rows x 16 K-pairs (+pad)
    __shared__ unsigned int Bu[64][17];    // 64 cols  x 16 K-pairs (+pad), B^T

    const int tid  = threadIdx.x;
    const int z    = blockIdx.z;
    const int zo   = z / innerN;
    const int zi   = z - zo * innerN;
    const float* Ab = A + (long long)zo * sAo + (long long)zi * sAi;
    const float* Bb = B + (long long)zo * sBo + (long long)zi * sBi;
    const float* Cb = Cres ? (Cres + (long long)zo * sCo + (long long)zi * sCi) : (const float*)0;
    float*       Db = D + (long long)zo * sDo + (long long)zi * sDi;

    const int m0 = blockIdx.y * 128;
    const int n0 = blockIdx.x * 64;
    const int wave = tid >> 5, lane = tid & 31;
    const int wm = (wave & 3) * 32;           // 0,32,64,96
    const int wn = (wave >> 2) * 32;          // 0,32
    const int ln = lane & 15;
    const int hi = lane >> 4;                 // 0 | 1
    const bool tb = (flags & FL_TB) != 0;

    f32x8 acc00 = {}, acc01 = {}, acc10 = {}, acc11 = {};
    const bool mnfull = (m0 + 128 <= M) && (n0 + 64 <= N);

    for (int k0 = 0; k0 < K; k0 += 32) {
        const bool full = mnfull && (k0 + 32 <= K);
        if (full) {
            // ---- A: 512 row-tasks of 8 consecutive K, 2 per thread ----
            #pragma unroll
            for (int t = 0; t < 2; t++) {
                int task = tid + t * 256;
                int r = task >> 2, kg = (task & 3) * 8;
                const float* s = Ab + (long long)(m0 + r) * lda + (k0 + kg);
                float v0 = s[0], v1 = s[1], v2 = s[2], v3 = s[3];
                float v4 = s[4], v5 = s[5], v6 = s[6], v7 = s[7];
                int kp = kg >> 1;
                Au[r][kp + 0] = pk2(v0, v1);
                Au[r][kp + 1] = pk2(v2, v3);
                Au[r][kp + 2] = pk2(v4, v5);
                Au[r][kp + 3] = pk2(v6, v7);
            }
            // ---- B: 256 col-tasks of 8 K, 1 per thread (store transposed) --
            {
                int r = tid >> 2, kg = (tid & 3) * 8;   // r = n-column
                float v[8];
                if (tb) {
                    const float* s = Bb + (long long)(n0 + r) * ldb + (k0 + kg);
                    #pragma unroll
                    for (int j = 0; j < 8; j++) v[j] = s[j];
                } else {
                    const float* s = Bb + (long long)(k0 + kg) * ldb + (n0 + r);
                    #pragma unroll
                    for (int j = 0; j < 8; j++) v[j] = s[(long long)j * ldb];
                }
                int kp = kg >> 1;
                Bu[r][kp + 0] = pk2(v[0], v[1]);
                Bu[r][kp + 1] = pk2(v[2], v[3]);
                Bu[r][kp + 2] = pk2(v[4], v[5]);
                Bu[r][kp + 3] = pk2(v[6], v[7]);
            }
        } else {
            // ---- guarded edge path ----
            #pragma unroll
            for (int t = 0; t < 2; t++) {
                int task = tid + t * 256;
                int r = task >> 2, kg = (task & 3) * 8;
                int gm = m0 + r;
                float v[8];
                #pragma unroll
                for (int j = 0; j < 8; j++) {
                    int gk = k0 + kg + j;
                    v[j] = (gm < M && gk < K) ? Ab[(long long)gm * lda + gk] : 0.f;
                }
                int kp = kg >> 1;
                Au[r][kp + 0] = pk2(v[0], v[1]);
                Au[r][kp + 1] = pk2(v[2], v[3]);
                Au[r][kp + 2] = pk2(v[4], v[5]);
                Au[r][kp + 3] = pk2(v[6], v[7]);
            }
            {
                int r = tid >> 2, kg = (tid & 3) * 8;
                int gn = n0 + r;
                float v[8];
                #pragma unroll
                for (int j = 0; j < 8; j++) {
                    int gk = k0 + kg + j;
                    float x = 0.f;
                    if (gn < N && gk < K)
                        x = tb ? Bb[(long long)gn * ldb + gk]
                               : Bb[(long long)gk * ldb + gn];
                    v[j] = x;
                }
                int kp = kg >> 1;
                Bu[r][kp + 0] = pk2(v[0], v[1]);
                Bu[r][kp + 1] = pk2(v[2], v[3]);
                Bu[r][kp + 2] = pk2(v[4], v[5]);
                Bu[r][kp + 3] = pk2(v[6], v[7]);
            }
        }
        __syncthreads();

        // ---- fragments: ISA 16-bit layout; u32 j holds elements 2j,2j+1 ->
        //      K-pair index kp = (j<4 ? j : j+4) + 4*(lane>=16)
        ABFrag a0, a1, b0, b1;
        #pragma unroll
        for (int j = 0; j < 8; j++) {
            int kp = ((j >> 2) << 3) + (j & 3) + (hi << 2);
            a0.u32[j] = Au[wm + ln][kp];
            a1.u32[j] = Au[wm + 16 + ln][kp];
            b0.u32[j] = Bu[wn + ln][kp];
            b1.u32[j] = Bu[wn + 16 + ln][kp];
        }
        acc00 = __builtin_amdgcn_wmma_f32_16x16x32_bf16(false, a0.v, false, b0.v,
                                                        (short)0, acc00, false, false);
        acc01 = __builtin_amdgcn_wmma_f32_16x16x32_bf16(false, a0.v, false, b1.v,
                                                        (short)0, acc01, false, false);
        acc10 = __builtin_amdgcn_wmma_f32_16x16x32_bf16(false, a1.v, false, b0.v,
                                                        (short)0, acc10, false, false);
        acc11 = __builtin_amdgcn_wmma_f32_16x16x32_bf16(false, a1.v, false, b1.v,
                                                        (short)0, acc11, false, false);
        __syncthreads();
    }

    const float alpha = alpha_ptr ? alpha_ptr[0] : 1.0f;
    #pragma unroll
    for (int ti = 0; ti < 2; ti++) {
        #pragma unroll
        for (int tj = 0; tj < 2; tj++) {
            #pragma unroll
            for (int r = 0; r < 8; r++) {
                int m = m0 + wm + ti * 16 + r + (hi << 3);  // C/D: vgpr r, hi-lanes M+8
                int n = n0 + wn + tj * 16 + ln;
                if (m < M && n < N) {
                    float v;
                    if (ti == 0 && tj == 0) v = acc00[r];
                    else if (ti == 0)       v = acc01[r];
                    else if (tj == 0)       v = acc10[r];
                    else                    v = acc11[r];
                    if (flags & FL_BIASN) v += bias[n];
                    if (flags & FL_BIASM) v += bias[m];
                    if (flags & FL_RELU)  v = v > 0.f ? v : 0.f;
                    if (flags & FL_GELU)  v = 0.5f * v * (1.f + erff(v * 0.70710678f));
                    v *= alpha;
                    if (flags & FL_RES)   v += Cb[(long long)m * ldc + n];
                    Db[(long long)m * ldd + n] = v;
                }
            }
        }
    }
}

// ---------------------------------------------------------------------------
// Direct NCHW conv (3x3 / 5x5), optional relu / residual add
// ---------------------------------------------------------------------------
__global__ void k_conv(const float* __restrict__ X, const float* __restrict__ Wt,
                       const float* __restrict__ bias, const float* __restrict__ Res,
                       float* __restrict__ Y,
                       int Bc, int Cin, int Cout, int H, int Wd, int Ksz, int pad,
                       int relu, int hasres)
{
    long long idx = (long long)blockIdx.x * 256 + threadIdx.x;
    long long total = (long long)Bc * Cout * H * Wd;
    if (idx >= total) return;
    int w  = (int)(idx % Wd);
    int h  = (int)((idx / Wd) % H);
    int co = (int)((idx / ((long long)Wd * H)) % Cout);
    int b  = (int)(idx / ((long long)Wd * H * Cout));
    float sum = bias[co];
    for (int ci = 0; ci < Cin; ci++) {
        const float* xp = X + (((long long)b * Cin + ci) * H) * Wd;
        const float* wp = Wt + (((long long)co * Cin + ci) * Ksz) * Ksz;
        for (int ky = 0; ky < Ksz; ky++) {
            int iy = h - pad + ky;
            if (iy < 0 || iy >= H) continue;
            for (int kx = 0; kx < Ksz; kx++) {
                int ix = w - pad + kx;
                if (ix < 0 || ix >= Wd) continue;
                sum += xp[(long long)iy * Wd + ix] * wp[ky * Ksz + kx];
            }
        }
    }
    if (relu) sum = sum > 0.f ? sum : 0.f;
    if (hasres) sum += Res[idx];
    Y[idx] = sum;
}

// ---------------------------------------------------------------------------
// DFT matrix generators (deterministic, regenerated every launch)
// ---------------------------------------------------------------------------
__global__ void k_dft128(float* cm, float* sm, float* smn,
                         float* icm, float* ism, float* ismn)
{
    int idx = blockIdx.x * 256 + threadIdx.x;
    if (idx >= 128 * 128) return;
    int w = idx >> 7, f = idx & 127;
    float th = 6.28318530717958647692f * (float)((w * f) & 127) / 128.f;
    float c = cosf(th), s = sinf(th);
    cm[idx] = c;          sm[idx] = -s;          smn[idx] = s;           // F = c - i s
    icm[idx] = c / 128.f; ism[idx] = s / 128.f;  ismn[idx] = -s / 128.f; // F^-1/N
}

__global__ void k_rfft256(float* wr, float* wi, float* ar, float* ai)
{
    int idx = blockIdx.x * 256 + threadIdx.x;
    if (idx >= 129 * 256) return;
    int f = idx / 256, s = idx & 255;
    float th = 6.28318530717958647692f * (float)((f * s) & 255) / 256.f;
    float c = cosf(th), sn = sinf(th);
    wr[idx] = c / 16.f;  wi[idx] = -sn / 16.f;                 // ortho rfft rows
    float cf = (f == 0 || f == 128) ? 1.f : 2.f;
    ar[(long long)s * 129 + f] =  cf * c  / 16.f;              // ortho irfft
    ai[(long long)s * 129 + f] = -cf * sn / 16.f;
}

// ---------------------------------------------------------------------------
// Elementwise helpers
// ---------------------------------------------------------------------------
__global__ void k_cabs(const float* zr, const float* zi, float* o, long long n) {
    long long i = (long long)blockIdx.x * 256 + threadIdx.x;
    if (i < n) o[i] = sqrtf(zr[i] * zr[i] + zi[i] * zi[i]);
}
__global__ void k_cangle(const float* zr, const float* zi, float* o, long long n) {
    long long i = (long long)blockIdx.x * 256 + threadIdx.x;
    if (i < n) o[i] = atan2f(zi[i], zr[i]);
}
__global__ void k_polar(const float* amp, const float* ph, float* re, float* im, long long n) {
    long long i = (long long)blockIdx.x * 256 + threadIdx.x;
    if (i < n) { float a = amp[i], p = ph[i]; re[i] = a * cosf(p); im[i] = a * sinf(p); }
}
__global__ void k_blend(const float* Xf, const float* Xs, const float* Wsig,
                        float* o, long long n, int hw2) {
    long long i = (long long)blockIdx.x * 256 + threadIdx.x;
    if (i >= n) return;
    int bc = (int)(i / hw2);
    float w = Wsig[bc];
    o[i] = Xf[i] * w + Xs[i] * (1.f - w);
}

// mean over H*W per (b,c)
__global__ __launch_bounds__(256) void k_meanhw(const float* X, float* Z, int hw2) {
    __shared__ float red[256];
    int bc = blockIdx.x, tid = threadIdx.x;
    const float* x = X + (long long)bc * hw2;
    float s = 0.f;
    for (int i = tid; i < hw2; i += 256) s += x[i];
    red[tid] = s; __syncthreads();
    for (int o = 128; o > 0; o >>= 1) { if (tid < o) red[tid] += red[tid + o]; __syncthreads(); }
    if (tid == 0) Z[bc] = red[0] / (float)hw2;
}

// squeeze-excite MLP: relu(z@fc1+b) -> sigmoid(@fc2+b). C=64, RED->4.
__global__ void k_se_fc(const float* z, const float* w1, const float* b1,
                        const float* w2, const float* b2, float* W) {
    __shared__ float zz[64], hh[4];
    int b = blockIdx.x, tid = threadIdx.x;       // 64 threads
    zz[tid] = z[b * 64 + tid]; __syncthreads();
    if (tid < 4) {
        float s = b1[tid];
        for (int c = 0; c < 64; c++) s += zz[c] * w1[c * 4 + tid];
        hh[tid] = s > 0.f ? s : 0.f;
    }
    __syncthreads();
    float s = b2[tid];
    for (int t = 0; t < 4; t++) s += hh[t] * w2[t * 64 + tid];
    W[b * 64 + tid] = 1.f / (1.f + expf(-s));
}

// im2col for stride==kernel patch conv
__global__ void k_im2col_patch(const float* x, float* col,
                               int Bc, int Cin, int H, int Wd, int P) {
    long long idx = (long long)blockIdx.x * 256 + threadIdx.x;
    int nh = H / P, nw = Wd / P, cpp = Cin * P * P;
    long long total = (long long)Bc * nh * nw * cpp;
    if (idx >= total) return;
    int cc  = (int)(idx % cpp);
    long long row = idx / cpp;
    int px = (int)(row % nw);
    int py = (int)((row / nw) % nh);
    int b  = (int)(row / ((long long)nw * nh));
    int c  = cc / (P * P);
    int iy = (cc / P) % P;
    int ix = cc % P;
    col[idx] = x[(((long long)b * Cin + c) * H + py * P + iy) * Wd + px * P + ix];
}

// fused = 0.5*(tok0[b,s] + tok1[b, s/4]) ; S=256, S1=64, D=768
__global__ void k_fuse_tokens(const float* t0, const float* t1, float* o) {
    long long idx = (long long)blockIdx.x * 256 + threadIdx.x;
    if (idx >= (long long)16 * 256 * 768) return;
    int d = (int)(idx % 768);
    int s = (int)((idx / 768) % 256);
    int b = (int)(idx / (768 * 256));
    float a = t0[((long long)b * 256 + s) * 768 + d];
    float c = t1[((long long)b * 64 + (s >> 2)) * 768 + d];
    o[idx] = 0.5f * (a + c);
}

// sinusoidal positional embedding add
__global__ void k_posadd(float* t) {
    long long idx = (long long)blockIdx.x * 256 + threadIdx.x;
    if (idx >= (long long)16 * 256 * 768) return;
    int d = (int)(idx % 768);
    int s = (int)((idx / 768) % 256);
    float div = expf((float)(d & ~1) * (-9.21034037198e0f / 768.f));
    float v = (d & 1) ? cosf((float)s * div) : sinf((float)s * div);
    t[idx] += v;
}

// row layernorm
__global__ __launch_bounds__(256) void k_layernorm(const float* X, const float* g,
                                                   const float* b, float* Y, int Dd) {
    __shared__ float red[256];
    int row = blockIdx.x, tid = threadIdx.x;
    const float* x = X + (long long)row * Dd;
    float s = 0.f;
    for (int i = tid; i < Dd; i += 256) s += x[i];
    red[tid] = s; __syncthreads();
    for (int o = 128; o > 0; o >>= 1) { if (tid < o) red[tid] += red[tid + o]; __syncthreads(); }
    float mean = red[0] / (float)Dd; __syncthreads();
    float v = 0.f;
    for (int i = tid; i < Dd; i += 256) { float d0 = x[i] - mean; v += d0 * d0; }
    red[tid] = v; __syncthreads();
    for (int o = 128; o > 0; o >>= 1) { if (tid < o) red[tid] += red[tid + o]; __syncthreads(); }
    float inv = rsqrtf(red[0] / (float)Dd + 1e-5f);
    for (int i = tid; i < Dd; i += 256)
        Y[(long long)row * Dd + i] = (x[i] - mean) * inv * g[i] + b[i];
}

// frequency attention weights: |qf * conj(kf)| = |qf||kf|, softmax over F=129,
// then O = V * w.   layout [bh][129][64]
__global__ void k_freqmix(const float* Qr, const float* Qi, const float* Kr, const float* Ki,
                          const float* Vr, const float* Vi, float* Or, float* Oi) {
    int bh = blockIdx.x, dc = threadIdx.x;   // 64 threads
    long long base = (long long)bh * 129 * 64 + dc;
    float mx = -1e30f;
    for (int f = 0; f < 129; f++) {
        long long p = base + (long long)f * 64;
        float qr = Qr[p], qi = Qi[p], kr = Kr[p], ki = Ki[p];
        float m = sqrtf(qr * qr + qi * qi) * sqrtf(kr * kr + ki * ki);
        Or[p] = m;
        mx = fmaxf(mx, m);
    }
    float s = 0.f;
    for (int f = 0; f < 129; f++) {
        long long p = base + (long long)f * 64;
        float e = expf(Or[p] - mx); Or[p] = e; s += e;
    }
    float inv = 1.f / s;
    for (int f = 0; f < 129; f++) {
        long long p = base + (long long)f * 64;
        float w = Or[p] * inv;
        Or[p] = Vr[p] * w; Oi[p] = Vi[p] * w;
    }
}

// mean over sequence: out[b,d] = mean_s t[b,s,d]
__global__ void k_meanS(const float* T, float* out) {
    int idx = blockIdx.x * 256 + threadIdx.x;
    if (idx >= 16 * 768) return;
    int b = idx / 768, d = idx % 768;
    float s = 0.f;
    for (int t = 0; t < 256; t++) s += T[((long long)b * 256 + t) * 768 + d];
    out[idx] = s / 256.f;
}

// ---------------------------------------------------------------------------
// Host-side launch plumbing
// ---------------------------------------------------------------------------
static void gemm(hipStream_t st,
                 const float* A, long long lda, long long sAo, long long sAi,
                 const float* B, long long ldb, long long sBo, long long sBi,
                 const float* bias, const float* Cres, long long ldc, long long sCo, long long sCi,
                 float* D, long long ldd, long long sDo, long long sDi,
                 const float* alpha, int M, int N, int K, int outer, int inner, int flags)
{
    dim3 g((unsigned)((N + 63) / 64), (unsigned)((M + 127) / 128), (unsigned)(outer * inner));
    k_gemm<<<g, 256, 0, st>>>(A, B, bias, Cres, D, alpha, M, N, K,
                              lda, ldb, ldc, ldd, sAo, sAi, sBo, sBi,
                              sCo, sCi, sDo, sDi, inner, flags);
}

extern "C" void kernel_launch(void* const* d_in, const int* in_sizes, int n_in,
                              void* d_out, int out_size, void* d_ws, size_t ws_size,
                              hipStream_t stream)
{
    (void)in_sizes; (void)n_in; (void)out_size; (void)ws_size;
    auto in = [&](int i) { return (const float*)d_in[i]; };

    // ---- input index map (recursive dict-insertion order) ----
    const float* X      = in(0);
    const float* sp_w = in(1),  *sp_b = in(2),  *fr_w = in(3),  *fr_b = in(4);
    const float* cc_w = in(5),  *cc_b = in(6),  *r1_w = in(7),  *r1_b = in(8);
    const float* r2_w = in(9),  *r2_b = in(10), *pc_w = in(11), *pc_b = in(12);
    const float* fc1w = in(13), *fc1b = in(14), *fc2w = in(15), *fc2b = in(16);
    const float* a1_w = in(17), *a1_b = in(18), *a2_w = in(19), *a2_b = in(20);
    const float* p1_w = in(21), *p1_b = in(22), *p2_w = in(23), *p2_b = in(24);
    const float* f_w  = in(25), *f_b  = in(26);
    const float* pj0w = in(27), *pj0b = in(28), *pj1w = in(29), *pj1b = in(30);
    const float* lng  = in(31), *lnb  = in(32);
    const int BLK0 = 33;           // 15 entries per transformer block
    const float* nfg = in(33 + 15 * 8), *nfb = in(34 + 15 * 8);

    // ---- workspace allocator ----
    size_t off = 0;
    auto alloc = [&](size_t n) { float* p = (float*)d_ws + off; off += n; return p; };
    const long long NIMG = 16LL * 64 * 128 * 128;     // 16,777,216
    const int HW2 = 128 * 128;

    float *CM = alloc(16384), *SM = alloc(16384), *SMN = alloc(16384);
    float *ICM = alloc(16384), *ISM = alloc(16384), *ISMN = alloc(16384);
    float *WR = alloc(129 * 256), *WI = alloc(129 * 256);
    float *AR = alloc(256 * 129), *AI = alloc(256 * 129);
    float *T1 = alloc(NIMG), *T2 = alloc(NIMG), *T3 = alloc(NIMG);
    float *T4 = alloc(NIMG), *T5 = alloc(NIMG), *T6 = alloc(NIMG);
    float *T7 = alloc(NIMG), *T8 = alloc(NIMG), *T9 = alloc(NIMG);
    float *zbuf = alloc(1024), *wsig = alloc(1024);
    float *tok0 = alloc(4096LL * 768), *tok1 = alloc(1024LL * 768);
    float *tA = alloc(4096LL * 768), *tB = alloc(4096LL * 768);
    float *tC = alloc(4096LL * 768), *tD = alloc(4096LL * 768);
    float *qb = alloc(4096LL * 768), *kb = alloc(4096LL * 768), *vb = alloc(4096LL * 768);
    float *tE = alloc(4096LL * 3072);
    const long long FQ = 192LL * 129 * 64;
    float *Qr = alloc(FQ), *Qi = alloc(FQ), *Kr = alloc(FQ), *Ki = alloc(FQ);
    float *Vr = alloc(FQ), *Vi = alloc(FQ), *Ofr = alloc(FQ), *Ofi = alloc(FQ);

    // ---- DFT matrices ----
    k_dft128<<<64, 256, 0, stream>>>(CM, SM, SMN, ICM, ISM, ISMN);
    k_rfft256<<<(129 * 256 + 255) / 256, 256, 0, stream>>>(WR, WI, AR, AI);

    const long long EW = (NIMG + 255) / 256;

    // helper: fft2 of real src -> (Zr=dr, Zi=di) using T1,T2 as scratch
    auto fft2 = [&](const float* src, float* dr, float* di) {
        // pass1 along W: Yr = X*CM, Yi = X*SM   (batched A, shared B)
        gemm(stream, src, 128, HW2, 0, CM, 128, 0, 0, 0, 0, 0, 0, 0,
             T1, 128, HW2, 0, 0, 128, 128, 128, 1024, 1, 0);
        gemm(stream, src, 128, HW2, 0, SM, 128, 0, 0, 0, 0, 0, 0, 0,
             T2, 128, HW2, 0, 0, 128, 128, 128, 1024, 1, 0);
        // pass2 along H: Zr = CM*Yr + SMN*Yi ; Zi = CM*Yi + SM*Yr
        gemm(stream, CM, 128, 0, 0, T1, 128, HW2, 0, 0, 0, 0, 0, 0,
             dr, 128, HW2, 0, 0, 128, 128, 128, 1024, 1, 0);
        gemm(stream, SMN, 128, 0, 0, T2, 128, HW2, 0, 0, dr, 128, HW2, 0,
             dr, 128, HW2, 0, 0, 128, 128, 128, 1024, 1, FL_RES);
        gemm(stream, CM, 128, 0, 0, T2, 128, HW2, 0, 0, 0, 0, 0, 0,
             di, 128, HW2, 0, 0, 128, 128, 128, 1024, 1, 0);
        gemm(stream, SM, 128, 0, 0, T1, 128, HW2, 0, 0, di, 128, HW2, 0,
             di, 128, HW2, 0, 0, 128, 128, 128, 1024, 1, FL_RES);
    };

    // ================= MDFM =================
    fft2(X, T3, T4);                                        // Zr=T3, Zi=T4
    k_cabs<<<EW, 256, 0, stream>>>(T3, T4, T5, NIMG);       // x_f = |fft2(x)|
    k_conv<<<EW, 256, 0, stream>>>(X, sp_w, sp_b, nullptr, T1, 16, 64, 64, 128, 128, 3, 1, 1, 0);  // Xs
    k_conv<<<EW, 256, 0, stream>>>(T5, fr_w, fr_b, nullptr, T2, 16, 64, 64, 128, 128, 5, 2, 1, 0); // Xf
    // cc 1x1 over concat [Xs, Xf]: T3 = ccW[:, :64]*Xs + ccW[:,64:]*Xf + b
    gemm(stream, cc_w, 128, 0, 0, T1, HW2, 64LL * HW2, 0, cc_b, 0, 0, 0, 0,
         T3, HW2, 64LL * HW2, 0, 0, 64, HW2, 64, 16, 1, FL_BIASM);
    gemm(stream, cc_w + 64, 128, 0, 0, T2, HW2, 64LL * HW2, 0, 0, T3, HW2, 64LL * HW2, 0,
         T3, HW2, 64LL * HW2, 0, 0, 64, HW2, 64, 16, 1, FL_RES);
    k_conv<<<EW, 256, 0, stream>>>(T3, r1_w, r1_b, nullptr, T4, 16, 64, 64, 128, 128, 3, 1, 1, 0);
    k_conv<<<EW, 256, 0, stream>>>(T4, r2_w, r2_b, T3, T5, 16, 64, 64, 128, 128, 3, 1, 0, 1);
    k_conv<<<EW, 256, 0, stream>>>(T5, pc_w, pc_b, nullptr, T4, 16, 64, 64, 128, 128, 3, 1, 0, 0);
    k_meanhw<<<1024, 256, 0, stream>>>(T4, zbuf, HW2);
    k_se_fc<<<16, 64, 0, stream>>>(zbuf, fc1w, fc1b, fc2w, fc2b, wsig);
    k_blend<<<EW, 256, 0, stream>>>(T2, T1, wsig, T6, NIMG, HW2);     // mdfm out = T6

    // ================= FFAB =================
    fft2(T6, T3, T4);
    k_cabs<<<EW, 256, 0, stream>>>(T3, T4, T5, NIMG);
    gemm(stream, a1_w, 64, 0, 0, T5, HW2, 64LL * HW2, 0, a1_b, 0, 0, 0, 0,
         T7, HW2, 64LL * HW2, 0, 0, 64, HW2, 64, 16, 1, FL_BIASM | FL_RELU);
    gemm(stream, a2_w, 64, 0, 0, T7, HW2, 64LL * HW2, 0, a2_b, 0, 0, 0, 0,
         T8, HW2, 64LL * HW2, 0, 0, 64, HW2, 64, 16, 1, FL_BIASM);           // amp = T8
    k_cangle<<<EW, 256, 0, stream>>>(T3, T4, T5, NIMG);
    gemm(stream, p1_w, 64, 0, 0, T5, HW2, 64LL * HW2, 0, p1_b, 0, 0, 0, 0,
         T7, HW2, 64LL * HW2, 0, 0, 64, HW2, 64, 16, 1, FL_BIASM | FL_RELU);
    gemm(stream, p2_w, 64, 0, 0, T7, HW2, 64LL * HW2, 0, p2_b, 0, 0, 0, 0,
         T9, HW2, 64LL * HW2, 0, 0, 64, HW2, 64, 16, 1, FL_BIASM);           // ph = T9
    k_polar<<<EW, 256, 0, stream>>>(T8, T9, T1, T2, NIMG);                   // Re=T1, Im=T2
    // ifft2, real part only: pass1 along W
    gemm(stream, T1, 128, HW2, 0, ICM, 128, 0, 0, 0, 0, 0, 0, 0,
         T3, 128, HW2, 0, 0, 128, 128, 128, 1024, 1, 0);
    gemm(stream, T2, 128, HW2, 0, ISMN, 128, 0, 0, 0, T3, 128, HW2, 0,
         T3, 128, HW2, 0, 0, 128, 128, 128, 1024, 1, FL_RES);
    gemm(stream, T1, 128, HW2, 0, ISM, 128, 0, 0, 0, 0, 0, 0, 0,
         T4, 128, HW2, 0, 0, 128, 128, 128, 1024, 1, 0);
    gemm(stream, T2, 128, HW2, 0, ICM, 128, 0, 0, 0, T4, 128, HW2, 0,
         T4, 128, HW2, 0, 0, 128, 128, 128, 1024, 1, FL_RES);
    // pass2 along H, real part: rec = ICM*Yr + ISMN*Yi
    gemm(stream, ICM, 128, 0, 0, T3, 128, HW2, 0, 0, 0, 0, 0, 0,
         T5, 128, HW2, 0, 0, 128, 128, 128, 1024, 1, 0);
    gemm(stream, ISMN, 128, 0, 0, T4, 128, HW2, 0, 0, T5, 128, HW2, 0,
         T5, 128, HW2, 0, 0, 128, 128, 128, 1024, 1, FL_RES);
    k_conv<<<EW, 256, 0, stream>>>(T5, f_w, f_b, T6, T7, 16, 64, 64, 128, 128, 3, 1, 0, 1); // y = T7

    // ================= Patch embed =================
    k_im2col_patch<<<EW, 256, 0, stream>>>(T7, T1, 16, 64, 128, 128, 8);    // 4096 x 4096
    k_im2col_patch<<<EW, 256, 0, stream>>>(T7, T2, 16, 64, 128, 128, 16);   // 1024 x 16384
    gemm(stream, T1, 4096, 0, 0, pj0w, 4096, 0, 0, pj0b, 0, 0, 0, 0,
         tok0, 768, 0, 0, 0, 4096, 768, 4096, 1, 1, FL_TB | FL_BIASN);
    gemm(stream, T2, 16384, 0, 0, pj1w, 16384, 0, 0, pj1b, 0, 0, 0, 0,
         tok1, 768, 0, 0, 0, 1024, 768, 16384, 1, 1, FL_TB | FL_BIASN);
    k_fuse_tokens<<<(4096 * 768 + 255) / 256, 256, 0, stream>>>(tok0, tok1, tA);
    k_layernorm<<<4096, 256, 0, stream>>>(tA, lng, lnb, tB, 768);
    k_posadd<<<(4096 * 768 + 255) / 256, 256, 0, stream>>>(tB);

    // ================= Transformer blocks =================
    for (int L = 0; L < 8; L++) {
        const float* wq = in(BLK0 + 15 * L + 0),  *bq = in(BLK0 + 15 * L + 1);
        const float* wk = in(BLK0 + 15 * L + 2),  *bk = in(BLK0 + 15 * L + 3);
        const float* wv = in(BLK0 + 15 * L + 4),  *bv = in(BLK0 + 15 * L + 5);
        const float* wo = in(BLK0 + 15 * L + 6),  *bo = in(BLK0 + 15 * L + 7);
        const float* ng = in(BLK0 + 15 * L + 8),  *nb = in(BLK0 + 15 * L + 9);
        const float* m1w = in(BLK0 + 15 * L + 10), *m1b = in(BLK0 + 15 * L + 11);
        const float* m2w = in(BLK0 + 15 * L + 12), *m2b = in(BLK0 + 15 * L + 13);
        const float* rs  = in(BLK0 + 15 * L + 14);

        k_layernorm<<<4096, 256, 0, stream>>>(tB, ng, nb, tC, 768);
        gemm(stream, tC, 768, 0, 0, wq, 768, 0, 0, bq, 0, 0, 0, 0,
             qb, 768, 0, 0, 0, 4096, 768, 768, 1, 1, FL_BIASN);
        gemm(stream, tC, 768, 0, 0, wk, 768, 0, 0, bk, 0, 0, 0, 0,
             kb, 768, 0, 0, 0, 4096, 768, 768, 1, 1, FL_BIASN);
        gemm(stream, tC, 768, 0, 0, wv, 768, 0, 0, bv, 0, 0, 0, 0,
             vb, 768, 0, 0, 0, 4096, 768, 768, 1, 1, FL_BIASN);

        // rfft over sequence per (b,h): Xf = W(129x256) * x(256x64), batch 16x12
        const long long sBo = 256LL * 768, sBi = 64;
        const long long sDo = 12LL * 129 * 64, sDi = 129LL * 64;
        struct { const float* src; float* fr; float* fi; } qkv[3] =
            { { qb, Qr, Qi }, { kb, Kr, Ki }, { vb, Vr, Vi } };
        for (int j = 0; j < 3; j++) {
            gemm(stream, WR, 256, 0, 0, qkv[j].src, 768, sBo, sBi, 0, 0, 0, 0, 0,
                 qkv[j].fr, 64, sDo, sDi, 0, 129, 64, 256, 16, 12, 0);
            gemm(stream, WI, 256, 0, 0, qkv[j].src, 768, sBo, sBi, 0, 0, 0, 0, 0,
                 qkv[j].fi, 64, sDo, sDi, 0, 129, 64, 256, 16, 12, 0);
        }
        k_freqmix<<<192, 64, 0, stream>>>(Qr, Qi, Kr, Ki, Vr, Vi, Ofr, Ofi);
        // irfft -> merged-head layout tD[b, s, h*64+d]
        gemm(stream, AR, 129, 0, 0, Ofr, 64, sDo, sDi, 0, 0, 0, 0, 0,
             tD, 768, 256LL * 768, 64, 0, 256, 64, 129, 16, 12, 0);
        gemm(stream, AI, 129, 0, 0, Ofi, 64, sDo, sDi, 0, tD, 768, 256LL * 768, 64,
             tD, 768, 256LL * 768, 64, 0, 256, 64, 129, 16, 12, FL_RES);
        // x = x + rs*(attn@wo + bo)
        gemm(stream, tD, 768, 0, 0, wo, 768, 0, 0, bo, tB, 768, 0, 0,
             tB, 768, 0, 0, rs, 4096, 768, 768, 1, 1, FL_BIASN | FL_RES);
        // MLP: h = gelu(x@m1+b1) ; x = x + rs*(h@m2+b2)
        gemm(stream, tB, 768, 0, 0, m1w, 3072, 0, 0, m1b, 0, 0, 0, 0,
             tE, 3072, 0, 0, 0, 4096, 3072, 768, 1, 1, FL_BIASN | FL_GELU);
        gemm(stream, tE, 3072, 0, 0, m2w, 768, 0, 0, m2b, tB, 768, 0, 0,
             tB, 768, 0, 0, rs, 4096, 768, 3072, 1, 1, FL_BIASN | FL_RES);
    }

    // ================= Head =================
    k_layernorm<<<4096, 256, 0, stream>>>(tB, nfg, nfb, tC, 768);
    k_meanS<<<(16 * 768 + 255) / 256, 256, 0, stream>>>(tC, (float*)d_out);
}
